// DualAttnModel_46883863003334
// MI455X (gfx1250) — compile-verified, matrix operand order checked by reading, not run on hardware
//
#include <hip/hip_runtime.h>
#include <hip/hip_bf16.h>

// ---------------------------------------------------------------------------
// DualAttnModel for MI455X (gfx1250, wave32, WMMA).
// B=32, P=16, L=256, H=256, N = B*P = 512 pages.
// Stage 1: gather emb -> X (f16, LDS) -> Q/K/V = X @ W^T + b via
//          v_wmma_f32_16x16x32_f16 (kk-outer, 8 live accumulators/wave).
// Stage 2: per (page, 32-query block): S = Q K^T / 16 (WMMA, K staged in LDS),
//          mask+softmax in LDS, O = P V (WMMA, V staged in LDS). Accumulate
//          query-means of P (word_attn) and of O (webpage) with f32 atomics.
// Stage 3: per page: residual = word_attn @ x (re-gather emb), LayerNorm.
// Stage 4: per batch: page scores, sparsemax(P=16), pooled LayerNorm, sigmoid.
// ---------------------------------------------------------------------------

typedef __attribute__((ext_vector_type(16))) _Float16 v16h;
typedef __attribute__((ext_vector_type(8)))  float    v8f;

#define LNUM   256   // tokens per page
#define HDIM   256   // hidden
#define NPAGE  512   // B*P
#define XLD    264   // padded LDS stride (halves) for 256-wide f16 tiles
#define SLD    264   // padded LDS stride (floats) for the score tile
#define KLD    40    // padded LDS stride (halves) for 32-wide K stage

// ---- WMMA fragment loaders (16x16x32 f16, wave32 layouts per ISA 7.12.2) ---
// A 16x32 (MxK): lanes 0-15 -> row M=lane, K = {0..7,16..23}; lanes 16-31 same
// rows, K = {8..15,24..31}.  16 halves per lane.
__device__ __forceinline__ v16h a_frag_f16(const _Float16* M, int ld,
                                           int mBase, int kBase, int lane) {
  const int m   = mBase + (lane & 15);
  const int khi = (lane >> 4) << 3;
  v16h a;
#pragma unroll
  for (int i = 0; i < 16; ++i) {
    const int k = kBase + khi + (i & 7) + ((i >> 3) << 4);
    a[i] = M[(size_t)m * ld + k];
  }
  return a;
}

__device__ __forceinline__ v16h a_frag_f32(const float* M, int ld,
                                           int mBase, int kBase, int lane) {
  const int m   = mBase + (lane & 15);
  const int khi = (lane >> 4) << 3;
  v16h a;
#pragma unroll
  for (int i = 0; i < 16; ++i) {
    const int k = kBase + khi + (i & 7) + ((i >> 3) << 4);
    a[i] = (_Float16)M[(size_t)m * ld + k];
  }
  return a;
}

// B 32x16 (KxN) from a row-major [N][K] matrix (B = W^T, W stored [n][k]).
__device__ __forceinline__ v16h b_frag_nk_f32(const float* W, int ld,
                                              int kBase, int nBase, int lane) {
  const int n   = nBase + (lane & 15);
  const int khi = (lane >> 4) << 3;
  v16h b;
#pragma unroll
  for (int i = 0; i < 16; ++i) {
    const int k = kBase + khi + (i & 7) + ((i >> 3) << 4);
    b[i] = (_Float16)W[(size_t)n * ld + k];
  }
  return b;
}

__device__ __forceinline__ v16h b_frag_nk_f16(const _Float16* W, int ld,
                                              int kBase, int nBase, int lane) {
  const int n   = nBase + (lane & 15);
  const int khi = (lane >> 4) << 3;
  v16h b;
#pragma unroll
  for (int i = 0; i < 16; ++i) {
    const int k = kBase + khi + (i & 7) + ((i >> 3) << 4);
    b[i] = W[(size_t)n * ld + k];
  }
  return b;
}

// B 32x16 (KxN) from a row-major [K][N] matrix (V staged [key][h] in LDS).
__device__ __forceinline__ v16h b_frag_kn_f16(const _Float16* M, int ld,
                                              int kBase, int nBase, int lane) {
  const int n   = nBase + (lane & 15);
  const int khi = (lane >> 4) << 3;
  v16h b;
#pragma unroll
  for (int i = 0; i < 16; ++i) {
    const int k = kBase + khi + (i & 7) + ((i >> 3) << 4);
    b[i] = M[(size_t)k * ld + n];
  }
  return b;
}

__device__ __forceinline__ v8f wmma_f16(v16h a, v16h b, v8f c) {
  return __builtin_amdgcn_wmma_f32_16x16x32_f16(false, a, false, b, (short)0,
                                                c, false, false);
}

// ---------------------------------------------------------------------------
// Stage 1: gather + QKV projection.  grid (rowBlocks=8, pages=512, mat=3),
// block 128 (4 waves).  kk-outer, each wave owns 2 m-tiles x 4 n-tiles.
// ---------------------------------------------------------------------------
__global__ __launch_bounds__(128) void qkv_kernel(
    const int* __restrict__ seq_ids, const float* __restrict__ emb,
    const float* __restrict__ q_w, const float* __restrict__ q_b,
    const float* __restrict__ k_w, const float* __restrict__ k_b,
    const float* __restrict__ v_w, const float* __restrict__ v_b,
    _Float16* __restrict__ Qh, _Float16* __restrict__ Kh,
    _Float16* __restrict__ Vh) {
  const int page = blockIdx.y;
  const int r0   = blockIdx.x * 32;
  const int mat  = blockIdx.z;

  const float* W;  const float* bias;  _Float16* Out;
  if (mat == 0)      { W = q_w; bias = q_b; Out = Qh; }
  else if (mat == 1) { W = k_w; bias = k_b; Out = Kh; }
  else               { W = v_w; bias = v_b; Out = Vh; }

  __shared__ _Float16 Xs[32 * XLD];

  for (int idx = threadIdx.x; idx < 32 * HDIM; idx += 128) {
    const int row = idx >> 8;
    const int col = idx & 255;
    const int tok = seq_ids[page * LNUM + r0 + row];
    Xs[row * XLD + col] = (_Float16)emb[(size_t)tok * HDIM + col];
  }
  __syncthreads();

  const int lane = threadIdx.x & 31;
  const int wave = threadIdx.x >> 5;

  v8f acc[2][4];
#pragma unroll
  for (int t = 0; t < 4; ++t) {
    const float bv = bias[(wave * 4 + t) * 16 + (lane & 15)];
#pragma unroll
    for (int r = 0; r < 8; ++r) { acc[0][t][r] = bv; acc[1][t][r] = bv; }
  }

  for (int kk = 0; kk < 8; ++kk) {
    v16h a0 = a_frag_f16(Xs, XLD, 0, kk * 32, lane);
    v16h a1 = a_frag_f16(Xs, XLD, 16, kk * 32, lane);
#pragma unroll
    for (int t = 0; t < 4; ++t) {
      v16h b = b_frag_nk_f32(W, HDIM, kk * 32, (wave * 4 + t) * 16, lane);
      acc[0][t] = wmma_f16(a0, b, acc[0][t]);
      acc[1][t] = wmma_f16(a1, b, acc[1][t]);
    }
  }

  const int mlane = (lane >> 4) << 3;
#pragma unroll
  for (int mt = 0; mt < 2; ++mt) {
#pragma unroll
    for (int t = 0; t < 4; ++t) {
      const int n = (wave * 4 + t) * 16 + (lane & 15);
#pragma unroll
      for (int r = 0; r < 8; ++r) {
        const int m = mt * 16 + mlane + r;
        Out[((size_t)(page * LNUM + r0 + m)) * HDIM + n] =
            (_Float16)acc[mt][t][r];
      }
    }
  }
}

// ---------------------------------------------------------------------------
// Stage 2: attention.  grid (qBlocks=8, pages=512), block 128 (4 waves).
// LDS: score tile S (33.8 KB) + KVs stage buffer (20.5 KB, K then V).
// ---------------------------------------------------------------------------
__global__ __launch_bounds__(128) void attn_kernel(
    const _Float16* __restrict__ Qh, const _Float16* __restrict__ Kh,
    const _Float16* __restrict__ Vh, const int* __restrict__ seq_lengths,
    float* __restrict__ word_accum, float* __restrict__ page_accum) {
  const int page = blockIdx.y;
  const int q0   = blockIdx.x * 32;
  const int lane = threadIdx.x & 31;
  const int wave = threadIdx.x >> 5;

  const _Float16* Qp = Qh + (size_t)page * LNUM * HDIM;
  const _Float16* Kp = Kh + (size_t)page * LNUM * HDIM;
  const _Float16* Vp = Vh + (size_t)page * LNUM * HDIM;

  __shared__ float    S[32 * SLD];       // 33792 B
  __shared__ _Float16 KVs[256 * KLD];    // 20480 B; phase1: K[256][KLD]
                                         //          phase3: V[32][XLD]

  // ---- phase 1: S = Q K^T / sqrt(H), K staged 32 h-cols at a time ----------
  {
    v8f acc[2][4];
#pragma unroll
    for (int t = 0; t < 4; ++t) { acc[0][t] = (v8f){}; acc[1][t] = (v8f){}; }

    for (int kk = 0; kk < 8; ++kk) {
      __syncthreads();
      // stage K[key=0..255][h = kk*32 .. kk*32+31] -> KVs[key][0..31]
      for (int idx = threadIdx.x; idx < 256 * 32; idx += 128) {
        const int row = idx >> 5;
        const int col = idx & 31;
        KVs[row * KLD + col] = Kp[(size_t)row * HDIM + kk * 32 + col];
      }
      __syncthreads();

      v16h a0 = a_frag_f16(Qp, HDIM, q0, kk * 32, lane);
      v16h a1 = a_frag_f16(Qp, HDIM, q0 + 16, kk * 32, lane);
#pragma unroll
      for (int t = 0; t < 4; ++t) {
        v16h b = b_frag_nk_f16(KVs, KLD, 0, (wave * 4 + t) * 16, lane);
        acc[0][t] = wmma_f16(a0, b, acc[0][t]);
        acc[1][t] = wmma_f16(a1, b, acc[1][t]);
      }
    }

    const int mlane = (lane >> 4) << 3;
#pragma unroll
    for (int mt = 0; mt < 2; ++mt) {
#pragma unroll
      for (int t = 0; t < 4; ++t) {
        const int n = (wave * 4 + t) * 16 + (lane & 15);
#pragma unroll
        for (int r = 0; r < 8; ++r)
          S[(mt * 16 + mlane + r) * SLD + n] = acc[mt][t][r] * 0.0625f;
      }
    }
  }
  __syncthreads();

  // ---- phase 2: mask (keys >= seq_length) + softmax over keys --------------
  const int seqlen = seq_lengths[page];
  if (threadIdx.x < 32) {
    float* row = S + threadIdx.x * SLD;
    float mx = -1e30f;
    for (int k = 0; k < LNUM; ++k) {
      const float v = (k >= seqlen) ? -1e30f : row[k];
      row[k] = v;
      mx = fmaxf(mx, v);
    }
    float sum = 0.0f;
    for (int k = 0; k < LNUM; ++k) {
      const float e = (row[k] <= -1e29f) ? 0.0f : __expf(row[k] - mx);
      row[k] = e;
      sum += e;
    }
    const float inv = 1.0f / sum;
    for (int k = 0; k < LNUM; ++k) row[k] *= inv;
  }
  __syncthreads();

  // column sums of attention probs -> word_attn accumulator
  for (int c = threadIdx.x; c < LNUM; c += 128) {
    float s = 0.0f;
#pragma unroll 4
    for (int r = 0; r < 32; ++r) s += S[r * SLD + c];
    atomicAdd(&word_accum[page * LNUM + c], s);
  }

  // ---- phase 3: O = P @ V, V staged 32 keys at a time ----------------------
  {
    v8f acc[2][4];
#pragma unroll
    for (int t = 0; t < 4; ++t) { acc[0][t] = (v8f){}; acc[1][t] = (v8f){}; }

    for (int kk = 0; kk < 8; ++kk) {
      __syncthreads();
      // stage V[key = kk*32 .. +31][h=0..255] -> KVs[key][h] (stride XLD)
      for (int idx = threadIdx.x; idx < 32 * HDIM; idx += 128) {
        const int row = idx >> 8;
        const int col = idx & 255;
        KVs[row * XLD + col] = Vp[(size_t)(kk * 32 + row) * HDIM + col];
      }
      __syncthreads();

      v16h a0 = a_frag_f32(S, SLD, 0, kk * 32, lane);
      v16h a1 = a_frag_f32(S, SLD, 16, kk * 32, lane);
#pragma unroll
      for (int t = 0; t < 4; ++t) {
        v16h b = b_frag_kn_f16(KVs, XLD, 0, (wave * 4 + t) * 16, lane);
        acc[0][t] = wmma_f16(a0, b, acc[0][t]);
        acc[1][t] = wmma_f16(a1, b, acc[1][t]);
      }
    }

#pragma unroll
    for (int mt = 0; mt < 2; ++mt) {
#pragma unroll
      for (int t = 0; t < 4; ++t) {
        float part = 0.0f;
#pragma unroll
        for (int r = 0; r < 8; ++r) part += acc[mt][t][r];
        atomicAdd(&page_accum[page * HDIM + (wave * 4 + t) * 16 + (lane & 15)],
                  part);
      }
    }
  }
}

// ---------------------------------------------------------------------------
// Stage 3: per-page residual + LayerNorm.  grid 512, block 256.
// ---------------------------------------------------------------------------
__global__ __launch_bounds__(256) void page_finalize_kernel(
    const float* __restrict__ word_accum, const float* __restrict__ page_accum,
    const int* __restrict__ seq_ids, const float* __restrict__ emb,
    const float* __restrict__ ln_g, const float* __restrict__ ln_b,
    float* __restrict__ out_word_attn, float* __restrict__ out_webpage) {
  const int page = blockIdx.x;
  const int h    = threadIdx.x;
  __shared__ float WA[256];
  __shared__ float red[256];

  const float wa = word_accum[page * LNUM + h] * (1.0f / 256.0f);
  out_word_attn[page * LNUM + h] = wa;
  WA[h] = wa;
  __syncthreads();

  float res = 0.0f;
  for (int l = 0; l < LNUM; ++l) {
    const int tok = seq_ids[page * LNUM + l];
    res += WA[l] * emb[(size_t)tok * HDIM + h];
  }
  const float v = page_accum[page * HDIM + h] * (1.0f / 256.0f) + res;

  red[h] = v;
  __syncthreads();
  for (int s = 128; s > 0; s >>= 1) { if (h < s) red[h] += red[h + s]; __syncthreads(); }
  const float mean = red[0] * (1.0f / 256.0f);
  __syncthreads();
  const float d = v - mean;
  red[h] = d * d;
  __syncthreads();
  for (int s = 128; s > 0; s >>= 1) { if (h < s) red[h] += red[h + s]; __syncthreads(); }
  const float var = red[0] * (1.0f / 256.0f);
  __syncthreads();

  out_webpage[page * HDIM + h] = d * rsqrtf(var + 1e-5f) * ln_g[h] + ln_b[h];
}

// ---------------------------------------------------------------------------
// Stage 4: page pooling (sparsemax) + final LayerNorm + decode. grid 32.
// ---------------------------------------------------------------------------
__global__ __launch_bounds__(256) void batch_finalize_kernel(
    const float* __restrict__ webpage, const int* __restrict__ num_pages,
    const float* __restrict__ W_page, const float* __restrict__ ln_g,
    const float* __restrict__ ln_b, const float* __restrict__ dec_w,
    float* __restrict__ out_probs, float* __restrict__ out_senti,
    float* __restrict__ out_page_attn, float* __restrict__ out_final,
    float* __restrict__ out_page_scores) {
  const int b = blockIdx.x;
  const int h = threadIdx.x;
  __shared__ float red[256];
  __shared__ float scores[16];
  __shared__ float attn[16];

  const float* wvb = webpage + (size_t)b * 16 * HDIM;
  const float wp = W_page[h];

  for (int p = 0; p < 16; ++p) {
    red[h] = wvb[p * HDIM + h] * wp;
    __syncthreads();
    for (int s = 128; s > 0; s >>= 1) { if (h < s) red[h] += red[h + s]; __syncthreads(); }
    if (h == 0) scores[p] = red[0] * 0.1f;  // / PAGE_SCALE
    __syncthreads();
  }
  if (h < 16) out_page_scores[b * 16 + h] = scores[h];

  const int np = num_pages[b];
  if (h == 0) {
    float z[16], zs[16];
    for (int p = 0; p < 16; ++p) { z[p] = (p >= np) ? -9999.0f : scores[p]; zs[p] = z[p]; }
    for (int i = 1; i < 16; ++i) {            // insertion sort, descending
      const float key = zs[i];
      int j = i - 1;
      while (j >= 0 && zs[j] < key) { zs[j + 1] = zs[j]; --j; }
      zs[j + 1] = key;
    }
    float cumarr[16];
    float c = 0.0f;
    int cnt = 0;
    for (int i = 0; i < 16; ++i) {
      c += zs[i];
      cumarr[i] = c;
      if (1.0f + (float)(i + 1) * zs[i] > c) ++cnt;
    }
    const float tau = (cumarr[cnt - 1] - 1.0f) / (float)cnt;
    for (int p = 0; p < 16; ++p) {
      attn[p] = fmaxf(z[p] - tau, 0.0f);
      out_page_attn[b * 16 + p] = attn[p];
    }
  }
  __syncthreads();

  float fv = 0.0f;
  for (int p = 0; p < 16; ++p) fv += attn[p] * wvb[p * HDIM + h];

  red[h] = fv;
  __syncthreads();
  for (int s = 128; s > 0; s >>= 1) { if (h < s) red[h] += red[h + s]; __syncthreads(); }
  const float mean = red[0] * (1.0f / 256.0f);
  __syncthreads();
  const float d = fv - mean;
  red[h] = d * d;
  __syncthreads();
  for (int s = 128; s > 0; s >>= 1) { if (h < s) red[h] += red[h + s]; __syncthreads(); }
  const float var = red[0] * (1.0f / 256.0f);
  __syncthreads();

  const float y = d * rsqrtf(var + 1e-5f) * ln_g[h] + ln_b[h];
  out_final[b * HDIM + h] = y;

  red[h] = y * dec_w[h];
  __syncthreads();
  for (int s = 128; s > 0; s >>= 1) { if (h < s) red[h] += red[h + s]; __syncthreads(); }
  if (h == 0) {
    const float s = red[0];
    out_senti[b] = s;
    out_probs[b] = 1.0f / (1.0f + __expf(-s));
  }
}

// ---------------------------------------------------------------------------
extern "C" void kernel_launch(void* const* d_in, const int* in_sizes, int n_in,
                              void* d_out, int out_size, void* d_ws,
                              size_t ws_size, hipStream_t stream) {
  const int*   seq_ids     = (const int*)d_in[0];
  const int*   num_pages   = (const int*)d_in[1];
  const int*   seq_lengths = (const int*)d_in[2];
  const float* emb         = (const float*)d_in[3];
  const float* q_w         = (const float*)d_in[4];
  const float* q_b         = (const float*)d_in[5];
  const float* k_w         = (const float*)d_in[6];
  const float* k_b         = (const float*)d_in[7];
  const float* v_w         = (const float*)d_in[8];
  const float* v_b         = (const float*)d_in[9];
  const float* ln_word_g   = (const float*)d_in[10];
  const float* ln_word_b   = (const float*)d_in[11];
  const float* W_page      = (const float*)d_in[12];
  const float* ln_page_g   = (const float*)d_in[13];
  const float* ln_page_b   = (const float*)d_in[14];
  const float* dec_w       = (const float*)d_in[15];

  // Output tuple layout (flat, f32): probs(32) | senti(32) | word_attn(131072)
  // | page_attn(512) | final_vec(8192) | page_scores(512) | webpage_vec(131072)
  float* out             = (float*)d_out;
  float* out_probs       = out;
  float* out_senti       = out + 32;
  float* out_word        = out + 64;
  float* out_page_attn   = out_word + 131072;
  float* out_final       = out_page_attn + 512;
  float* out_page_scores = out_final + 8192;
  float* out_webpage     = out_page_scores + 512;

  // Workspace: Qh/Kh/Vh f16 (64 MB each) + two f32 accumulators (512 KB each)
  char*  ws       = (char*)d_ws;
  size_t qkvBytes = (size_t)NPAGE * LNUM * HDIM * sizeof(_Float16);
  _Float16* Qh = (_Float16*)ws;
  _Float16* Kh = (_Float16*)(ws + qkvBytes);
  _Float16* Vh = (_Float16*)(ws + 2 * qkvBytes);
  float* page_accum = (float*)(ws + 3 * qkvBytes);
  float* word_accum = (float*)(ws + 3 * qkvBytes + (size_t)NPAGE * HDIM * sizeof(float));

  hipMemsetAsync(page_accum, 0, 2 * (size_t)NPAGE * HDIM * sizeof(float), stream);

  qkv_kernel<<<dim3(8, NPAGE, 3), 128, 0, stream>>>(
      seq_ids, emb, q_w, q_b, k_w, k_b, v_w, v_b, Qh, Kh, Vh);
  attn_kernel<<<dim3(8, NPAGE), 128, 0, stream>>>(
      Qh, Kh, Vh, seq_lengths, word_accum, page_accum);
  page_finalize_kernel<<<NPAGE, 256, 0, stream>>>(
      word_accum, page_accum, seq_ids, emb, ln_word_g, ln_word_b, out_word,
      out_webpage);
  batch_finalize_kernel<<<32, 256, 0, stream>>>(
      out_webpage, num_pages, W_page, ln_page_g, ln_page_b, dec_w, out_probs,
      out_senti, out_page_attn, out_final, out_page_scores);
}
